// MultiHeadCrossAttention_6451040878908
// MI455X (gfx1250) — matrix operand order repaired and checked by reading, not compiled, not run on hardware
//
#include <hip/hip_runtime.h>
#include <stdint.h>

// ---- problem constants ----
#define BATCH   4
#define NKVTOT  2048
#define NQTOT   1024
#define DMODEL  1024
#define NHEAD   16
#define HDIM    64

typedef __attribute__((ext_vector_type(16))) _Float16 v16h;
typedef __attribute__((ext_vector_type(8)))  _Float16 v8h;
typedef __attribute__((ext_vector_type(8)))  float    v8f;

union V16 { v16h v; v8h h[2]; };

// Load 8 contiguous halves, converting from f32 on the fly if needed.
static __device__ __forceinline__ v8h load8h(const _Float16* p) {
    return *(const v8h*)p;
}
static __device__ __forceinline__ v8h load8h(const float* p) {
    float4 a = *(const float4*)p;
    float4 b = *(const float4*)(p + 4);
    v8h r = { (_Float16)a.x, (_Float16)a.y, (_Float16)a.z, (_Float16)a.w,
              (_Float16)b.x, (_Float16)b.y, (_Float16)b.z, (_Float16)b.w };
    return r;
}

static __device__ __forceinline__ v8f wmma_f16(v16h a, v16h b, v8f c) {
    return __builtin_amdgcn_wmma_f32_16x16x32_f16(false, a, false, b,
                                                  (short)0, c, false, false);
}

// Async global->LDS copy of 16B per lane (gfx1250 GLOBAL_LOAD_ASYNC_TO_LDS_B128,
// tracked by ASYNCcnt). IOFFSET adds to both the LDS and global address.
static __device__ __forceinline__ void async_copy16(void* lds_ptr, const void* gptr) {
    const uint32_t lds_off = (uint32_t)(uintptr_t)lds_ptr;   // LDS address from low 32 bits
    asm volatile("global_load_async_to_lds_b128 %0, %1, off"
                 :: "v"(lds_off), "v"(gptr) : "memory");
}
static __device__ __forceinline__ void async_copy32(void* lds_ptr, const void* gptr) {
    const uint32_t lds_off = (uint32_t)(uintptr_t)lds_ptr;
    asm volatile("global_load_async_to_lds_b128 %0, %1, off"
                 :: "v"(lds_off), "v"(gptr) : "memory");
    asm volatile("global_load_async_to_lds_b128 %0, %1, off offset:16"
                 :: "v"(lds_off), "v"(gptr) : "memory");
}
static __device__ __forceinline__ void wait_async0() {
    asm volatile("s_wait_asynccnt 0x0" ::: "memory");
}

// ---------------------------------------------------------------------------
// Tiled GEMM: C[M,N] = A[M,K] * B[K,N] + bias[N]
// A: row-major (f32 or f16), B: row-major (f32 or f16), C: f16 or f32.
// 128x128 tile / 256 threads (8 waves); wave -> 32x64 subtile (2x4 WMMA tiles).
// f16 A tiles are staged with async global->LDS DMA (no VGPR bounce).
// ---------------------------------------------------------------------------
template <typename AT, typename BT, typename CT>
__global__ __launch_bounds__(256)
void gemm_tile(const AT* __restrict__ A, const BT* __restrict__ B,
               const float* __restrict__ bias, CT* __restrict__ C,
               int M, int N, int K) {
    constexpr int BM = 128, BN = 128, BK = 32;
    constexpr int LDA = BK + 8;   // 40 halves = 80B row stride (16B-multiple)
    constexpr int LDB = BK + 8;
    __shared__ _Float16 As[BM * LDA];
    __shared__ _Float16 Bs[BN * LDB];   // stored transposed: [n][k]

    const int tid  = threadIdx.x;
    const int lane = tid & 31;
    const int wave = tid >> 5;
    const int wm   = wave & 3;          // 4 waves along M
    const int wn   = wave >> 2;         // 2 waves along N
    const int l15  = lane & 15;
    const int lhi  = lane >> 4;
    const int m0   = blockIdx.y * BM;
    const int n0   = blockIdx.x * BN;

    v8f acc[2][4];
#pragma unroll
    for (int i = 0; i < 2; ++i)
#pragma unroll
        for (int j = 0; j < 4; ++j) acc[i][j] = (v8f){0,0,0,0,0,0,0,0};

    const int ar = tid >> 2;            // A: 64 rows/pass, 4 thr * 8 halves
    const int ak = (tid & 3) * 8;
    const int bk = tid >> 3;            // B: 32 k rows, 8 thr * 16 n each
    const int bn = (tid & 7) * 16;

    for (int k0 = 0; k0 < K; k0 += BK) {
        // --- stage A tile ---
        if (sizeof(AT) == 2) {
            // f16 source: async DMA straight into LDS (layout-preserving)
#pragma unroll
            for (int rr = ar; rr < BM; rr += 64)
                async_copy16(&As[rr * LDA + ak],
                             A + (size_t)(m0 + rr) * K + k0 + ak);
        } else {
#pragma unroll
            for (int rr = ar; rr < BM; rr += 64) {
                v8h va = load8h(A + (size_t)(m0 + rr) * K + k0 + ak);
                *(v8h*)(&As[rr * LDA + ak]) = va;
            }
        }
        // --- stage B tile transposed: Bs[n][k] (converts f32->f16) ---
        {
            const BT* gb = B + (size_t)(k0 + bk) * N + n0 + bn;
            v8h b0 = load8h(gb);
            v8h b1 = load8h(gb + 8);
#pragma unroll
            for (int i = 0; i < 8; ++i) Bs[(bn + i) * LDB + bk]     = b0[i];
#pragma unroll
            for (int i = 0; i < 8; ++i) Bs[(bn + 8 + i) * LDB + bk] = b1[i];
        }
        if (k0 + BK < K) {  // prefetch next slab -> global_prefetch_b8
            __builtin_prefetch(A + (size_t)(m0 + ar) * K + k0 + BK + ak, 0, 1);
            __builtin_prefetch(B + (size_t)(k0 + BK + bk) * N + n0 + bn, 0, 1);
        }
        if (sizeof(AT) == 2) wait_async0();
        __syncthreads();

        // --- fragments + WMMA ---
        v16h af[2], bf[4];
#pragma unroll
        for (int i = 0; i < 2; ++i) {
            const _Float16* p = &As[(wm * 32 + i * 16 + l15) * LDA + lhi * 8];
            V16 u; u.h[0] = *(const v8h*)p; u.h[1] = *(const v8h*)(p + 16);
            af[i] = u.v;
        }
#pragma unroll
        for (int j = 0; j < 4; ++j) {
            const _Float16* p = &Bs[(wn * 64 + j * 16 + l15) * LDB + lhi * 8];
            V16 u; u.h[0] = *(const v8h*)p; u.h[1] = *(const v8h*)(p + 16);
            bf[j] = u.v;
        }
#pragma unroll
        for (int i = 0; i < 2; ++i)
#pragma unroll
            for (int j = 0; j < 4; ++j)
                acc[i][j] = wmma_f16(af[i], bf[j], acc[i][j]);
        __syncthreads();
    }

    // --- epilogue: bias + store (C layout: VGPR e -> row e (+8 hi half)) ---
#pragma unroll
    for (int j = 0; j < 4; ++j) {
        const int col = n0 + wn * 64 + j * 16 + l15;
        const float bv = bias[col];
#pragma unroll
        for (int i = 0; i < 2; ++i)
#pragma unroll
            for (int e = 0; e < 8; ++e) {
                const int row = m0 + wm * 32 + i * 16 + e + lhi * 8;
                C[(size_t)row * N + col] = (CT)(acc[i][j][e] + bv);
            }
    }
}

// ---------------------------------------------------------------------------
// Mask packer: 1 bit per (b,q,kv) element via wave ballot.
// ---------------------------------------------------------------------------
__global__ __launch_bounds__(256)
void pack_mask(const int* __restrict__ mask, uint32_t* __restrict__ pm) {
    const int tid = blockIdx.x * blockDim.x + threadIdx.x;
    const unsigned long long bal = __ballot(mask[tid] != 0);
    if ((threadIdx.x & 31) == 0) pm[tid >> 5] = (uint32_t)bal;
}

// ---------------------------------------------------------------------------
// Flash attention: one wave per 16 q rows; stream 64-wide KV tiles via LDS.
// kvh: [b][kv][h*128 + d] (k d<64, v d>=64) f16.  qh: [b][q][h*64+d] f16.
// K tile staged with async global->LDS DMA; V tile transposed via VGPRs.
// ---------------------------------------------------------------------------
__global__ __launch_bounds__(256)
void attention_kernel(const _Float16* __restrict__ kvh,
                      const _Float16* __restrict__ qh,
                      const uint32_t* __restrict__ pm,
                      _Float16* __restrict__ aoh) {
    constexpr int LK = HDIM + 8;             // 72 halves = 144B (16B-multiple)
    __shared__ _Float16 Kt[64 * LK];         // [kv][hd]
    __shared__ _Float16 Vt[64 * LK];         // transposed: [hd][kv]
    __shared__ _Float16 Ps[8 * 16 * LK];     // per-wave P (16 x 64)

    const int tid  = threadIdx.x;
    const int lane = tid & 31;
    const int wave = tid >> 5;
    const int l15  = lane & 15;
    const int lhi  = lane >> 4;
    const int b    = blockIdx.y / NHEAD;
    const int h    = blockIdx.y % NHEAD;
    const int q0   = blockIdx.x * 128 + wave * 16;

    // Q fragments (A-layout), resident for the whole KV loop
    v16h qa[2];
    {
        const _Float16* qp = qh + (size_t)(b * NQTOT + q0 + l15) * DMODEL + h * HDIM;
#pragma unroll
        for (int s = 0; s < 2; ++s) {
            const _Float16* p = qp + s * 32 + lhi * 8;
            V16 u; u.h[0] = *(const v8h*)p; u.h[1] = *(const v8h*)(p + 16);
            qa[s] = u.v;
        }
    }

    float mrow[8], lrow[8];
    v8f oacc[4];
#pragma unroll
    for (int e = 0; e < 8; ++e) { mrow[e] = -__builtin_inff(); lrow[e] = 0.f; }
#pragma unroll
    for (int t = 0; t < 4; ++t) oacc[t] = (v8f){0,0,0,0,0,0,0,0};

    _Float16* Pw = &Ps[wave * 16 * LK];
    const int kr = tid >> 2;                 // KV stage: 64 rows, 4 thr * 16
    const int kc = (tid & 3) * 16;

    for (int kt = 0; kt < NKVTOT / 64; ++kt) {
        const int kv0 = kt * 64;
        // --- stage K via async DMA, V transposed via VGPRs (overlapped) ---
        {
            const _Float16* kp = kvh + (size_t)(b * NKVTOT + kv0 + kr) * (2 * DMODEL)
                                     + h * (2 * HDIM);
            async_copy32(&Kt[kr * LK + kc], kp + kc);   // K rows, layout-preserving
            v8h v0v = *(const v8h*)(kp + HDIM + kc);
            v8h v1v = *(const v8h*)(kp + HDIM + kc + 8);
#pragma unroll
            for (int i = 0; i < 8; ++i) Vt[(kc + i) * LK + kr]     = v0v[i];
#pragma unroll
            for (int i = 0; i < 8; ++i) Vt[(kc + 8 + i) * LK + kr] = v1v[i];
            wait_async0();
        }
        __syncthreads();

        // --- S = Q K^T (4 n-tiles of 16 kv, 2 K-slices of 32) ---
        v8f sacc[4];
#pragma unroll
        for (int t = 0; t < 4; ++t) sacc[t] = (v8f){0,0,0,0,0,0,0,0};
#pragma unroll
        for (int s = 0; s < 2; ++s)
#pragma unroll
            for (int t = 0; t < 4; ++t) {
                const _Float16* p = &Kt[(t * 16 + l15) * LK + s * 32 + lhi * 8];
                V16 u; u.h[0] = *(const v8h*)p; u.h[1] = *(const v8h*)(p + 16);
                sacc[t] = wmma_f16(qa[s], u.v, sacc[t]);
            }

        // --- scale + mask (packed bits) ---
        float sc[4][8];
#pragma unroll
        for (int e = 0; e < 8; ++e) {
            const int qr = q0 + e + lhi * 8;
            const uint32_t* pr = pm + ((size_t)b * NQTOT + qr) * (NKVTOT / 32) + (kv0 >> 5);
            const uint32_t w0 = pr[0], w1 = pr[1];  // covers kv0 .. kv0+63
#pragma unroll
            for (int t = 0; t < 4; ++t) {
                const uint32_t w  = (t < 2) ? w0 : w1;
                const uint32_t bit = (w >> ((t & 1) * 16 + l15)) & 1u;
                const float v = sacc[t][e] * 0.125f;   // 1/sqrt(64)
                sc[t][e] = bit ? v : -1e20f;
            }
        }

        // --- online softmax (rows span 16-lane half-groups) ---
#pragma unroll
        for (int e = 0; e < 8; ++e) {
            float cm = sc[0][e];
#pragma unroll
            for (int t = 1; t < 4; ++t) cm = fmaxf(cm, sc[t][e]);
#pragma unroll
            for (int off = 1; off < 16; off <<= 1)
                cm = fmaxf(cm, __shfl_xor(cm, off, 32));
            const float nm = fmaxf(mrow[e], cm);
            const float scale = __expf(mrow[e] - nm);
            float rs = 0.f;
#pragma unroll
            for (int t = 0; t < 4; ++t) {
                const float p = __expf(sc[t][e] - nm);
                sc[t][e] = p;
                rs += p;
            }
#pragma unroll
            for (int off = 1; off < 16; off <<= 1)
                rs += __shfl_xor(rs, off, 32);
            lrow[e] = lrow[e] * scale + rs;
            mrow[e] = nm;
#pragma unroll
            for (int t = 0; t < 4; ++t) oacc[t][e] = oacc[t][e] * scale;
        }

        // --- P -> LDS (C-layout to A-layout bounce; same-wave, DS in-order) ---
#pragma unroll
        for (int t = 0; t < 4; ++t)
#pragma unroll
            for (int e = 0; e < 8; ++e)
                Pw[(e + lhi * 8) * LK + t * 16 + l15] = (_Float16)sc[t][e];

        // --- O += P V ---
#pragma unroll
        for (int s = 0; s < 2; ++s) {
            const _Float16* pp = &Pw[l15 * LK + s * 32 + lhi * 8];
            V16 u; u.h[0] = *(const v8h*)pp; u.h[1] = *(const v8h*)(pp + 16);
            const v16h pa = u.v;
#pragma unroll
            for (int t = 0; t < 4; ++t) {
                const _Float16* vp = &Vt[(t * 16 + l15) * LK + s * 32 + lhi * 8];
                V16 w; w.h[0] = *(const v8h*)vp; w.h[1] = *(const v8h*)(vp + 16);
                oacc[t] = wmma_f16(pa, w.v, oacc[t]);
            }
        }
        __syncthreads();
    }

    // --- normalize + store f16 ---
#pragma unroll
    for (int e = 0; e < 8; ++e) {
        const float inv = 1.f / lrow[e];
        const int qr = q0 + e + lhi * 8;
        _Float16* op = aoh + (size_t)(b * NQTOT + qr) * DMODEL + h * HDIM;
#pragma unroll
        for (int t = 0; t < 4; ++t)
            op[t * 16 + l15] = (_Float16)(oacc[t][e] * inv);
    }
}

// ---------------------------------------------------------------------------
extern "C" void kernel_launch(void* const* d_in, const int* in_sizes, int n_in,
                              void* d_out, int out_size, void* d_ws, size_t ws_size,
                              hipStream_t stream) {
    const float* x    = (const float*)d_in[0];
    const float* y    = (const float*)d_in[1];
    const int*   mask = (const int*)  d_in[2];
    const float* Wkv  = (const float*)d_in[3];
    const float* bkv  = (const float*)d_in[4];
    const float* Wq   = (const float*)d_in[5];
    const float* bq   = (const float*)d_in[6];
    const float* Wo   = (const float*)d_in[7];
    const float* bo   = (const float*)d_in[8];
    float* out = (float*)d_out;

    // workspace layout (f16 regions, then packed mask)
    _Float16* kvh = (_Float16*)d_ws;                                   // 8192 x 2048
    _Float16* qh  = kvh + (size_t)BATCH * NKVTOT * 2 * DMODEL;         // 4096 x 1024
    _Float16* aoh = qh  + (size_t)BATCH * NQTOT * DMODEL;              // 4096 x 1024
    uint32_t* pm  = (uint32_t*)(aoh + (size_t)BATCH * NQTOT * DMODEL); // 262144 words

    const dim3 blk(256);

    // mask -> bitmask (1 bit/elem): 8.4M elems
    pack_mask<<<(BATCH * NQTOT * NKVTOT) / 256, blk, 0, stream>>>(mask, pm);

    // kv = x @ Wkv + bkv  (M=8192, N=2048, K=1024), f16 out
    gemm_tile<float, float, _Float16>
        <<<dim3(2048 / 128, 8192 / 128), blk, 0, stream>>>(x, Wkv, bkv, kvh,
                                                           8192, 2048, 1024);
    // q = y @ Wq + bq  (M=4096, N=1024, K=1024), f16 out
    gemm_tile<float, float, _Float16>
        <<<dim3(1024 / 128, 4096 / 128), blk, 0, stream>>>(y, Wq, bq, qh,
                                                           4096, 1024, 1024);
    // attention: grid (q-blocks, b*h)
    attention_kernel<<<dim3(NQTOT / 128, BATCH * NHEAD), blk, 0, stream>>>(
        kvh, qh, pm, aoh);

    // out = ao @ Wo + bo  (M=4096, N=1024, K=1024), f32 out
    gemm_tile<_Float16, float, float>
        <<<dim3(1024 / 128, 4096 / 128), blk, 0, stream>>>(aoh, Wo, bo, out,
                                                           4096, 1024, 1024);
}
